// VIBO_1PL_44659069944377
// MI455X (gfx1250) — matrix operand order repaired and checked by reading, not compiled, not run on hardware
//
#include <hip/hip_runtime.h>
#include <cmath>

typedef _Float16 v16h __attribute__((ext_vector_type(16)));
typedef _Float16 h8   __attribute__((ext_vector_type(8)));
typedef float    v8f  __attribute__((ext_vector_type(8)));

union V16 { v16h v; h8 h[2]; };

#define P_N 4096
#define I_N 256
#define H_N 64
#define A_N 32

// Branchless ELU: max(x,0) + exp(min(x,0)) - 1  (exp(0)==1 exactly).
// Uses the hardware v_exp_f32 transcendental, no divergent branch.
__device__ __forceinline__ float eluf(float x) {
    return fmaxf(x, 0.0f) + __expf(fminf(x, 0.0f)) - 1.0f;
}

__device__ __forceinline__ v8f wmma_f16(const V16& a, const V16& b, v8f c) {
    return __builtin_amdgcn_wmma_f32_16x16x32_f16(false, a.v, false, b.v,
                                                  (short)0, c, false, false);
}

// ---------------------------------------------------------------------------
// Kernel 1: fused encoder MLP + masked mean over items. One block per person.
// 4 waves; wave w owns output columns [16w, 16w+16).
// ---------------------------------------------------------------------------
__global__ __launch_bounds__(128) void k1_encoder(
    const float* __restrict__ response, const int* __restrict__ mask,
    const float* __restrict__ w1, const float* __restrict__ b1,
    const float* __restrict__ w2, const float* __restrict__ b2,
    float* __restrict__ hid_mean)
{
    __shared__ __align__(16) _Float16 h1s[I_N * H_N];  // 32 KB: h1, item-major
    __shared__ __align__(16) _Float16 w2h[H_N * H_N];  // 8 KB:  w2 in f16
    __shared__ float maskf[I_N];
    __shared__ float s_denom;

    const int p   = blockIdx.x;
    const int tid = threadIdx.x;

    if (tid == 0) s_denom = 0.0f;
    __syncthreads();

    // masks + denom
    float mlocal = 0.0f;
    for (int i = tid; i < I_N; i += 128) {
        float mv = (float)mask[p * I_N + i];
        maskf[i] = mv;
        mlocal += mv;
    }
    atomicAdd(&s_denom, mlocal);

    // w2 -> f16 LDS
    for (int idx = tid; idx < H_N * H_N; idx += 128)
        w2h[idx] = (_Float16)w2[idx];

    // h1[i,j] = elu(r_i * w1[j] + b1[j])  (w1 is (1,H))
    for (int i = tid; i < I_N; i += 128) {
        const float r = response[p * I_N + i];
        #pragma unroll 8
        for (int j = 0; j < H_N; ++j) {
            float v = fmaf(r, w1[j], b1[j]);
            h1s[i * H_N + j] = (_Float16)eluf(v);
        }
    }
    __syncthreads();

    const float denom = fmaxf(s_denom, 1.0f);

    const int lane = tid & 31;
    const int wv   = tid >> 5;        // N-tile id
    const int nl   = lane & 15;
    const int hi   = lane >> 4;
    const int ng   = wv * 16 + nl;    // global output column j'

    // B fragments (w2 columns), K-halves 0..31 and 32..63.
    // B 32x16 layout: lane (nl,hi), element j -> K = hi*8 + j + (j & 8).
    V16 b0, b1f;
    #pragma unroll
    for (int j = 0; j < 16; ++j) {
        b0.v[j]  = w2h[(hi * 8 + j + (j & 8)) * H_N + ng];
        b1f.v[j] = w2h[(32 + hi * 8 + j + (j & 8)) * H_N + ng];
    }
    const float bias2 = b2[ng];

    float asum = 0.0f;
    const int base = hi * 8;

    for (int it = 0; it < 16; ++it) {
        // A 16x32 layout: lane row m = nl; elems 0..7 -> K=base..base+7,
        // elems 8..15 -> K=base+16..base+23 (+32 for second K-half).
        const _Float16* rowp = &h1s[(it * 16 + nl) * H_N];
        V16 a0, a1;
        a0.h[0] = *(const h8*)(rowp + base);
        a0.h[1] = *(const h8*)(rowp + base + 16);
        a1.h[0] = *(const h8*)(rowp + base + 32);
        a1.h[1] = *(const h8*)(rowp + base + 48);

        v8f c = {};
        c = wmma_f16(a0, b0, c);
        c = wmma_f16(a1, b1f, c);

        // C/D: lane column n = nl, VGPR v -> row m = v + hi*8
        #pragma unroll
        for (int v = 0; v < 8; ++v) {
            const int item = it * 16 + hi * 8 + v;
            float val = eluf(c[v] + bias2);
            asum = fmaf(val, maskf[item], asum);
        }
    }

    // columns are duplicated across lane L and L^16: combine halves
    asum += __shfl_xor(asum, 16, 32);
    if (hi == 0)
        hid_mean[p * H_N + ng] = asum / denom;
}

// ---------------------------------------------------------------------------
// Kernel 2: decoder MLP (w3 -> elu -> w4) for 16 persons per block + ability
// reparameterization sums. 4 waves; wave w owns columns [16w, 16w+16).
// ---------------------------------------------------------------------------
__global__ __launch_bounds__(128) void k2_decoder(
    const float* __restrict__ hid_mean,
    const float* __restrict__ w3, const float* __restrict__ b3,
    const float* __restrict__ w4, const float* __restrict__ b4,
    const float* __restrict__ eps_ability,
    float* __restrict__ out_mu, float* __restrict__ out_lv,
    float* __restrict__ abil_sum)
{
    __shared__ __align__(16) _Float16 a_lds[16 * H_N];
    __shared__ __align__(16) _Float16 t_lds[16 * H_N];
    __shared__ float out_lds[16 * H_N];

    const int tid  = threadIdx.x;
    const int p0   = blockIdx.x * 16;
    const int lane = tid & 31;
    const int wv   = tid >> 5;
    const int nl   = lane & 15;
    const int hi   = lane >> 4;
    const int ng   = wv * 16 + nl;
    const int base = hi * 8;

    for (int idx = tid; idx < 16 * H_N; idx += 128)
        a_lds[idx] = (_Float16)hid_mean[p0 * H_N + idx];
    __syncthreads();

    // ---- layer 3: t = elu(hid_mean @ w3 + b3) ----
    V16 b0, b1f;
    #pragma unroll
    for (int j = 0; j < 16; ++j) {
        b0.v[j]  = (_Float16)w3[(hi * 8 + j + (j & 8)) * H_N + ng];
        b1f.v[j] = (_Float16)w3[(32 + hi * 8 + j + (j & 8)) * H_N + ng];
    }
    {
        const _Float16* rowp = &a_lds[nl * H_N];
        V16 a0, a1;
        a0.h[0] = *(const h8*)(rowp + base);
        a0.h[1] = *(const h8*)(rowp + base + 16);
        a1.h[0] = *(const h8*)(rowp + base + 32);
        a1.h[1] = *(const h8*)(rowp + base + 48);
        v8f c = {};
        c = wmma_f16(a0, b0, c);
        c = wmma_f16(a1, b1f, c);
        const float bias3 = b3[ng];
        #pragma unroll
        for (int v = 0; v < 8; ++v)
            t_lds[(hi * 8 + v) * H_N + ng] = (_Float16)eluf(c[v] + bias3);
    }
    __syncthreads();

    // ---- layer 4: out = t @ w4 + b4 ----
    #pragma unroll
    for (int j = 0; j < 16; ++j) {
        b0.v[j]  = (_Float16)w4[(hi * 8 + j + (j & 8)) * H_N + ng];
        b1f.v[j] = (_Float16)w4[(32 + hi * 8 + j + (j & 8)) * H_N + ng];
    }
    {
        const _Float16* rowp = &t_lds[nl * H_N];
        V16 a0, a1;
        a0.h[0] = *(const h8*)(rowp + base);
        a0.h[1] = *(const h8*)(rowp + base + 16);
        a1.h[0] = *(const h8*)(rowp + base + 32);
        a1.h[1] = *(const h8*)(rowp + base + 48);
        v8f c = {};
        c = wmma_f16(a0, b0, c);
        c = wmma_f16(a1, b1f, c);
        const float bias4 = b4[ng];
        #pragma unroll
        for (int v = 0; v < 8; ++v) {
            const int row = hi * 8 + v;
            const float o = c[v] + bias4;
            out_lds[row * H_N + ng] = o;
            const int pp = p0 + row;
            if (ng < A_N) out_mu[pp * A_N + ng] = o;
            else          out_lv[pp * A_N + (ng - A_N)] = o;
        }
    }
    __syncthreads();

    // ability sums: sum_a (mu + eps * exp(0.5*logvar))
    if (tid < 16) {
        const int pp = p0 + tid;
        float s = 0.0f;
        for (int a = 0; a < A_N; ++a) {
            const float mu = out_lds[tid * H_N + a];
            const float lv = out_lds[tid * H_N + A_N + a];
            s = fmaf(eps_ability[pp * A_N + a], __expf(0.5f * lv), s + mu);
        }
        abil_sum[pp] = s;
    }
}

// ---------------------------------------------------------------------------
// Kernel 3: rank-1 sigmoid outer product + item outputs. Bandwidth-bound.
// ---------------------------------------------------------------------------
__global__ __launch_bounds__(256) void k3_response(
    const float* __restrict__ abil_sum,
    const int* __restrict__ item_index,
    const float* __restrict__ eps_item,
    const float* __restrict__ mu_table, const float* __restrict__ logvar_table,
    float* __restrict__ resp_mu,
    float* __restrict__ out_item_mu, float* __restrict__ out_item_lv)
{
    const int p = blockIdx.x;
    const int i = threadIdx.x;
    const int idx = item_index[i];
    const float imu = mu_table[idx];
    const float ilv = logvar_table[idx];
    const float feat = fmaf(eps_item[i], __expf(0.5f * ilv), imu);
    const float logit = abil_sum[p] + feat;
    resp_mu[p * I_N + i] = 1.0f / (1.0f + __expf(-logit));
    if (p == 0) {
        out_item_mu[i] = imu;
        out_item_lv[i] = ilv;
    }
}

extern "C" void kernel_launch(void* const* d_in, const int* in_sizes, int n_in,
                              void* d_out, int out_size, void* d_ws, size_t ws_size,
                              hipStream_t stream) {
    const float* response     = (const float*)d_in[0];
    const int*   mask         = (const int*)d_in[1];
    const int*   item_index   = (const int*)d_in[2];
    const float* eps_ability  = (const float*)d_in[3];
    const float* eps_item     = (const float*)d_in[4];
    const float* w1 = (const float*)d_in[5];
    const float* b1 = (const float*)d_in[6];
    const float* w2 = (const float*)d_in[7];
    const float* b2 = (const float*)d_in[8];
    const float* w3 = (const float*)d_in[9];
    const float* b3 = (const float*)d_in[10];
    const float* w4 = (const float*)d_in[11];
    const float* b4 = (const float*)d_in[12];
    const float* mu_table     = (const float*)d_in[13];
    const float* logvar_table = (const float*)d_in[14];

    float* out     = (float*)d_out;
    float* resp_mu = out;                    // P*I
    float* o_mu    = resp_mu + P_N * I_N;    // P*A
    float* o_lv    = o_mu + P_N * A_N;       // P*A
    float* o_imu   = o_lv + P_N * A_N;       // I
    float* o_ilv   = o_imu + I_N;            // I

    float* hid_mean = (float*)d_ws;          // P*H floats
    float* abil_sum = hid_mean + P_N * H_N;  // P floats

    k1_encoder<<<P_N, 128, 0, stream>>>(response, mask, w1, b1, w2, b2, hid_mean);
    k2_decoder<<<P_N / 16, 128, 0, stream>>>(hid_mean, w3, b3, w4, b4,
                                             eps_ability, o_mu, o_lv, abil_sum);
    k3_response<<<P_N, I_N, 0, stream>>>(abil_sum, item_index, eps_item,
                                         mu_table, logvar_table,
                                         resp_mu, o_imu, o_ilv);
}
